// CGCNN_NO_BN_64046552318041
// MI455X (gfx1250) — matrix operand order repaired and checked by reading, not compiled, not run on hardware
//
#include <hip/hip_runtime.h>
#include <hip/hip_bf16.h>

// ---------------------------------------------------------------------------
// CGCNN (no BN) for MI455X / gfx1250, wave32, WMMA f32_16x16x32_f16
// ---------------------------------------------------------------------------

#define N_NODES  50000
#define N_EDGES  800000
#define NODE_D   128
#define EDGE_D   100
#define EMB_D    92
#define N_GRAPHS 256
#define KTOT     384      // 128 (xi) + 128 (xj) + 128 (ea padded from 100)
#define MT       32       // edges (M rows) per block
#define ASTR     392      // LDS A-tile row stride in halves (bank-stagger pad)
#define ZSTR     260      // LDS zc row stride in floats (bank-stagger pad)

typedef _Float16 half_t;
typedef __attribute__((ext_vector_type(16))) _Float16 v16h;
typedef __attribute__((ext_vector_type(8)))  float    v8f;

union Frag16 { v16h v; uint4 q[2]; };

__device__ __forceinline__ float softplus_f(float x) {
    return (x > 20.0f) ? x : log1pf(__expf(x));
}
__device__ __forceinline__ float sigmoid_f(float x) {
    return 1.0f / (1.0f + __expf(-x));
}

// ---------------------------------------------------------------------------
// Edge attributes: d = |R[src]-R[dst]|, gaussian smearing -> fp16, padded 128
// block = 256 threads = 8 waves, one edge per wave
// ---------------------------------------------------------------------------
__global__ __launch_bounds__(256) void edge_attr_kernel(
        const float* __restrict__ R, const int* __restrict__ ei,
        half_t* __restrict__ ea16) {
    const int tid  = threadIdx.x;
    const int e    = blockIdx.x * 8 + (tid >> 5);
    const int lane = tid & 31;
    const int s = ei[e];
    const int d = ei[N_EDGES + e];
    const float dx = R[3 * s + 0] - R[3 * d + 0];
    const float dy = R[3 * s + 1] - R[3 * d + 1];
    const float dz = R[3 * s + 2] - R[3 * d + 2];
    const float dist = sqrtf(dx * dx + dy * dy + dz * dz);
    const float step  = 6.0f / 99.0f;
    const float coeff = -0.5f / (step * step);
    union { half_t h[4]; uint2 u; } pk;
#pragma unroll
    for (int j = 0; j < 4; ++j) {
        const int k = lane * 4 + j;
        float v = 0.0f;
        if (k < EDGE_D) { const float t = dist - (float)k * step; v = __expf(coeff * t * t); }
        pk.h[j] = (half_t)v;
    }
    *(uint2*)&ea16[(size_t)e * 128 + lane * 4] = pk.u;
}

// ---------------------------------------------------------------------------
// x = embedding[z] @ emb_w + emb_b    (one node per 128-thread block)
// ---------------------------------------------------------------------------
__global__ __launch_bounds__(128) void embed_kernel(
        const int* __restrict__ z, const float* __restrict__ emb,
        const float* __restrict__ ew, const float* __restrict__ eb,
        float* __restrict__ x) {
    __shared__ float row[EMB_D];
    const int n = blockIdx.x;
    const int t = threadIdx.x;
    if (t < EMB_D) row[t] = emb[(size_t)z[n] * EMB_D + t];
    __syncthreads();
    float acc = eb[t];
#pragma unroll 4
    for (int k = 0; k < EMB_D; ++k) acc += row[k] * ew[k * NODE_D + t];
    x[(size_t)n * NODE_D + t] = acc;
}

// ---------------------------------------------------------------------------
// Pack conv weights: Wt[c][k] = (k<356) ? conv_w[k][c] : 0    fp16 [256][384]
// ---------------------------------------------------------------------------
__global__ __launch_bounds__(256) void pack_w_kernel(
        const float* __restrict__ cw, half_t* __restrict__ Wt) {
    const int idx = blockIdx.x * 256 + threadIdx.x;   // grid 384 -> 98304 = 256*384
    const int c = idx / KTOT;
    const int k = idx - c * KTOT;
    Wt[idx] = (k < 356) ? (half_t)cw[(size_t)k * 256 + c] : (half_t)0.0f;
}

__global__ __launch_bounds__(256) void zero_f32(float* __restrict__ p, int count) {
    const int i = blockIdx.x * 256 + threadIdx.x;
    if (i < count) p[i] = 0.0f;
}

// ---------------------------------------------------------------------------
// Fused edge-conv GEMM + gate + scatter-add.
// Block: 256 threads (8 waves), MT=32 edges. Each wave: 32x32 output patch
// (2 M-tiles x 2 N-tiles, 4 accumulators; B fragments shared across M-tiles).
// zc[e,:256] = [x[dst] | x[src] | ea] @ Wt^T ; msg = sig(z1+b1)*softplus(z2+b2)
// atomicAdd into agg[dst].
//
// Fragment layouts (CDNA5 ISA 7.12.2, f16 16x16x32):
//   A (16x32, row per lane):  m = lane&15, lanes 0-15: K {0..7,16..23},
//                             lanes 16-31: K {8..15,24..31}
//   B (32x16, col per lane):  n = lane&15, lanes 0-15: K {0..7,16..23},
//                             lanes 16-31: K {8..15,24..31}
//   C/D (16x16 f32): VGPR v -> row v + 8*(lane>>4), col lane&15
// ---------------------------------------------------------------------------
__global__ __launch_bounds__(256) void conv_gemm_scatter(
        const float* __restrict__ x, const half_t* __restrict__ ea16,
        const half_t* __restrict__ Wt, const float* __restrict__ bias,
        const int* __restrict__ ei, float* __restrict__ agg) {
    __shared__ half_t Af[MT * ASTR];     // 25088 B
    __shared__ float  zc[MT * ZSTR];     // 33280 B
    __shared__ int    dstIdx[MT], srcIdx[MT];

    const int tid = threadIdx.x;
    const int eb  = blockIdx.x * MT;

    if (tid < MT) {
        srcIdx[tid] = ei[eb + tid];
        dstIdx[tid] = ei[N_EDGES + eb + tid];
    }
    // Pull the (L2-resident) weight panel toward the WGP while we gather A.
    // 8 waves x 32 lanes covers all 256 rows of Wt.
    __builtin_prefetch(Wt + (size_t)tid * KTOT, 0, 2);
    __syncthreads();

    // Gather x[dst]|x[src] (f32 -> f16) : MT rows x 256 cols, 4 floats/thread/iter
    for (int i = tid * 4; i < MT * 256; i += 256 * 4) {
        const int m = i >> 8;
        const int c = i & 255;
        const float* sp = (c < 128) ? (x + (size_t)dstIdx[m] * NODE_D + c)
                                    : (x + (size_t)srcIdx[m] * NODE_D + (c - 128));
        const float4 v = *(const float4*)sp;
        union { half_t h[4]; uint2 u; } pk;
        pk.h[0] = (half_t)v.x; pk.h[1] = (half_t)v.y;
        pk.h[2] = (half_t)v.z; pk.h[3] = (half_t)v.w;
        *(uint2*)&Af[m * ASTR + c] = pk.u;
    }
    // Edge-attr slab: MT rows x 128 halves, 8 halves (16B) per thread
    for (int i = tid * 8; i < MT * 128; i += 256 * 8) {
        const int m = i >> 7;
        const int c = i & 127;
        *(uint4*)&Af[m * ASTR + 256 + c] =
            *(const uint4*)&ea16[(size_t)(eb + m) * 128 + c];
    }
    __syncthreads();

    const int lane  = tid & 31;
    const int wave  = tid >> 5;
    const int nlan  = lane & 15;
    const int hs    = lane >> 4;
    const int koffA = hs * 8;    // A fragment K-sublane offset (halves)
    const int koffB = hs * 8;    // B fragment K-sublane offset (halves)
    const int n0    = wave * 32;

    v8f acc00 = {}, acc01 = {}, acc10 = {}, acc11 = {};
    const half_t* w0  = Wt + (size_t)(n0 + nlan) * KTOT;
    const half_t* w1  = w0 + 16 * KTOT;
    const half_t* ar0 = &Af[nlan * ASTR];
    const half_t* ar1 = &Af[(16 + nlan) * ASTR];

#pragma unroll
    for (int ks = 0; ks < KTOT / 32; ++ks) {
        const int kb = ks * 32;
        Frag16 a0, a1, b0, b1;
        a0.q[0] = *(const uint4*)(ar0 + kb + koffA);
        a0.q[1] = *(const uint4*)(ar0 + kb + 16 + koffA);
        a1.q[0] = *(const uint4*)(ar1 + kb + koffA);
        a1.q[1] = *(const uint4*)(ar1 + kb + 16 + koffA);
        b0.q[0] = *(const uint4*)(w0 + kb + koffB);
        b0.q[1] = *(const uint4*)(w0 + kb + koffB + 16);
        b1.q[0] = *(const uint4*)(w1 + kb + koffB);
        b1.q[1] = *(const uint4*)(w1 + kb + koffB + 16);
        acc00 = __builtin_amdgcn_wmma_f32_16x16x32_f16(false, a0.v, false, b0.v,
                                                       (short)0, acc00, false, false);
        acc01 = __builtin_amdgcn_wmma_f32_16x16x32_f16(false, a0.v, false, b1.v,
                                                       (short)0, acc01, false, false);
        acc10 = __builtin_amdgcn_wmma_f32_16x16x32_f16(false, a1.v, false, b0.v,
                                                       (short)0, acc10, false, false);
        acc11 = __builtin_amdgcn_wmma_f32_16x16x32_f16(false, a1.v, false, b1.v,
                                                       (short)0, acc11, false, false);
    }

    // Spill D tiles to LDS using documented C/D layout: row = v + 8*(lane>>4)
#pragma unroll
    for (int v = 0; v < 8; ++v) {
        const int m = v + 8 * hs;
        zc[m * ZSTR + n0 + nlan]             = acc00[v];
        zc[m * ZSTR + n0 + 16 + nlan]        = acc01[v];
        zc[(16 + m) * ZSTR + n0 + nlan]      = acc10[v];
        zc[(16 + m) * ZSTR + n0 + 16 + nlan] = acc11[v];
    }
    __syncthreads();

    // Gate + scatter: msg[m,f] = sigmoid(z1)*softplus(z2) -> atomicAdd agg[dst]
    for (int i = tid; i < MT * NODE_D; i += 256) {
        const int m = i >> 7;
        const int f = i & 127;
        const float z1 = zc[m * ZSTR + f]       + bias[f];
        const float z2 = zc[m * ZSTR + 128 + f] + bias[128 + f];
        const float msg = sigmoid_f(z1) * softplus_f(z2);
        atomicAdd(&agg[(size_t)dstIdx[m] * NODE_D + f], msg);
    }
}

// ---------------------------------------------------------------------------
// x = softplus(layernorm(agg)*g + b + x)   one node per wave
// ---------------------------------------------------------------------------
__global__ __launch_bounds__(256) void node_update_kernel(
        const float* __restrict__ agg, const float* __restrict__ lg,
        const float* __restrict__ lb, float* __restrict__ x) {
    const int tid  = threadIdx.x;
    const int n    = blockIdx.x * 8 + (tid >> 5);
    const int lane = tid & 31;
    const float4 a  = *(const float4*)&agg[(size_t)n * NODE_D + lane * 4];
    const float4 xo = *(const float4*)&x[(size_t)n * NODE_D + lane * 4];
    const float av[4] = {a.x, a.y, a.z, a.w};
    const float xv[4] = {xo.x, xo.y, xo.z, xo.w};
    float s  = av[0] + av[1] + av[2] + av[3];
    float ss = av[0]*av[0] + av[1]*av[1] + av[2]*av[2] + av[3]*av[3];
    for (int m = 16; m > 0; m >>= 1) {
        s  += __shfl_xor(s,  m, 32);
        ss += __shfl_xor(ss, m, 32);
    }
    const float mean = s * (1.0f / 128.0f);
    float var = ss * (1.0f / 128.0f) - mean * mean;
    var = fmaxf(var, 0.0f);
    const float rstd = rsqrtf(var + 1e-5f);
    float out[4];
#pragma unroll
    for (int j = 0; j < 4; ++j) {
        const int c = lane * 4 + j;
        const float ln = (av[j] - mean) * rstd * lg[c] + lb[c];
        out[j] = softplus_f(ln + xv[j]);
    }
    *(float4*)&x[(size_t)n * NODE_D + lane * 4] = *(float4*)out;
}

// ---------------------------------------------------------------------------
// Global mean pool
// ---------------------------------------------------------------------------
__global__ __launch_bounds__(256) void pool_atomic_kernel(
        const float* __restrict__ x, const int* __restrict__ batch,
        float* __restrict__ sums, float* __restrict__ cnts) {
    const int idx = blockIdx.x * 256 + threadIdx.x;   // N_NODES*128
    const int n = idx >> 7;
    const int f = idx & 127;
    const int g = batch[n];
    atomicAdd(&sums[(size_t)g * NODE_D + f], x[idx]);
    if (f == 0) atomicAdd(&cnts[g], 1.0f);
}

__global__ __launch_bounds__(128) void pool_div_kernel(
        const float* __restrict__ sums, const float* __restrict__ cnts,
        float* __restrict__ mol) {
    const int idx = blockIdx.x * 128 + threadIdx.x;   // 256*128
    const int g = idx >> 7;
    mol[idx] = sums[idx] / fmaxf(cnts[g], 1.0f);
}

// ---------------------------------------------------------------------------
// Tiny MLP layers: out[g,f] = softplus(in[g,:] @ W[:,f] + b[f])
// ---------------------------------------------------------------------------
__global__ __launch_bounds__(128) void mlp_kernel(
        const float* __restrict__ in, const float* __restrict__ W,
        const float* __restrict__ b, float* __restrict__ out) {
    __shared__ float row[128];
    const int g = blockIdx.x;
    const int f = threadIdx.x;
    row[f] = in[(size_t)g * 128 + f];
    __syncthreads();
    float acc = b[f];
#pragma unroll 4
    for (int k = 0; k < 128; ++k) acc += row[k] * W[k * 128 + f];
    out[(size_t)g * 128 + f] = softplus_f(acc);
}

__global__ __launch_bounds__(128) void out_kernel(
        const float* __restrict__ h, const float* __restrict__ ow,
        const float* __restrict__ ob, float* __restrict__ out) {
    __shared__ float red[128];
    const int g = blockIdx.x;
    const int t = threadIdx.x;
    red[t] = h[(size_t)g * 128 + t] * ow[t];
    __syncthreads();
    for (int s = 64; s > 0; s >>= 1) {
        if (t < s) red[t] += red[t + s];
        __syncthreads();
    }
    if (t == 0) out[g] = red[0] + ob[0];
}

// ---------------------------------------------------------------------------
extern "C" void kernel_launch(void* const* d_in, const int* in_sizes, int n_in,
                              void* d_out, int out_size, void* d_ws, size_t ws_size,
                              hipStream_t stream) {
    const int*   z      = (const int*)  d_in[0];
    const float* R      = (const float*)d_in[1];
    const int*   ei     = (const int*)  d_in[2];
    const int*   batch  = (const int*)  d_in[3];
    const float* emb    = (const float*)d_in[4];
    const float* emb_w  = (const float*)d_in[5];
    const float* emb_b  = (const float*)d_in[6];
    const float* conv_w = (const float*)d_in[7];
    const float* conv_b = (const float*)d_in[8];
    const float* ln_g   = (const float*)d_in[9];
    const float* ln_b   = (const float*)d_in[10];
    const float* cfc_w  = (const float*)d_in[11];
    const float* cfc_b  = (const float*)d_in[12];
    const float* fc_w   = (const float*)d_in[13];
    const float* fc_b   = (const float*)d_in[14];
    const float* out_w  = (const float*)d_in[15];
    const float* out_b  = (const float*)d_in[16];
    float* out = (float*)d_out;

    char* wsp = (char*)d_ws;
    auto take = [&](size_t bytes) -> char* {
        char* p = wsp;
        wsp += (bytes + 255) & ~(size_t)255;
        return p;
    };
    half_t* ea16 = (half_t*)take((size_t)N_EDGES * 128 * sizeof(half_t)); // 204.8 MB
    float*  x    = (float*) take((size_t)N_NODES * NODE_D * sizeof(float));
    float*  agg  = (float*) take((size_t)N_NODES * NODE_D * sizeof(float));
    half_t* Wt   = (half_t*)take((size_t)256 * KTOT * sizeof(half_t));
    float*  sums = (float*) take((size_t)N_GRAPHS * NODE_D * sizeof(float));
    float*  cnts = (float*) take((size_t)N_GRAPHS * sizeof(float));
    float*  hA   = (float*) take((size_t)N_GRAPHS * 128 * sizeof(float));
    float*  hB   = (float*) take((size_t)N_GRAPHS * 128 * sizeof(float));

    edge_attr_kernel<<<N_EDGES / 8, 256, 0, stream>>>(R, ei, ea16);
    embed_kernel<<<N_NODES, 128, 0, stream>>>(z, emb, emb_w, emb_b, x);

    for (int l = 0; l < 3; ++l) {
        pack_w_kernel<<<KTOT, 256, 0, stream>>>(conv_w + (size_t)l * 356 * 256, Wt);
        zero_f32<<<(N_NODES * NODE_D + 255) / 256, 256, 0, stream>>>(agg, N_NODES * NODE_D);
        conv_gemm_scatter<<<N_EDGES / MT, 256, 0, stream>>>(
            x, ea16, Wt, conv_b + (size_t)l * 256, ei, agg);
        node_update_kernel<<<N_NODES / 8, 256, 0, stream>>>(
            agg, ln_g + (size_t)l * NODE_D, ln_b + (size_t)l * NODE_D, x);
    }

    zero_f32<<<(N_GRAPHS * NODE_D + 255) / 256, 256, 0, stream>>>(sums, N_GRAPHS * NODE_D);
    zero_f32<<<1, 256, 0, stream>>>(cnts, N_GRAPHS);
    pool_atomic_kernel<<<(N_NODES * NODE_D) / 256, 256, 0, stream>>>(x, batch, sums, cnts);
    pool_div_kernel<<<N_GRAPHS, 128, 0, stream>>>(sums, cnts, hA);   // hA = mol

    mlp_kernel<<<N_GRAPHS, 128, 0, stream>>>(hA, cfc_w, cfc_b, hB);
    mlp_kernel<<<N_GRAPHS, 128, 0, stream>>>(hB, fc_w, fc_b, hA);
    mlp_kernel<<<N_GRAPHS, 128, 0, stream>>>(hA, fc_w + 128 * 128, fc_b + 128, hB);
    out_kernel<<<N_GRAPHS, 128, 0, stream>>>(hB, out_w, out_b, out);
}
